// Net_38620345926088
// MI455X (gfx1250) — matrix operand order repaired and checked by reading, not compiled, not run on hardware
//
#include <hip/hip_runtime.h>
#include <hip/hip_bf16.h>
#include <math.h>

// ---------------------------------------------------------------------------
// Problem constants (from reference)
// ---------------------------------------------------------------------------
#define NN      20000
#define EE      320000
#define ETOT    (EE + NN)        // edges + self loops = 340000
#define F_IN    50
#define KPAD1   64               // layer-1 K padded to 64 (zeros)
#define HID     64
#define HEADS   8
#define WIDTH   (HID * HEADS)    // 512
#define NCLS    121

typedef __attribute__((ext_vector_type(16))) _Float16 v16h;
typedef __attribute__((ext_vector_type(8)))  float    v8f;

// ---------------------------------------------------------------------------
// f32 -> f16 conversions
// ---------------------------------------------------------------------------
// row-major copy with K -> Kpad zero padding: out[n][k<Kpad] from in[n][k<K]
__global__ void cvt_pad_rows(const float* __restrict__ in, _Float16* __restrict__ out,
                             int rows, int K, int Kpad) {
    long long t = (long long)blockIdx.x * blockDim.x + threadIdx.x;
    if (t >= (long long)rows * Kpad) return;
    int n = (int)(t / Kpad), k = (int)(t - (long long)n * Kpad);
    out[t] = (k < K) ? (_Float16)in[(long long)n * K + k] : (_Float16)0.0f;
}

// transpose + convert + pad: in[K][N] row-major -> out[N][Kpad] (weights)
__global__ void cvt_T_pad(const float* __restrict__ in, _Float16* __restrict__ out,
                          int K, int N, int Kpad) {
    long long t = (long long)blockIdx.x * blockDim.x + threadIdx.x;
    if (t >= (long long)N * Kpad) return;
    int n = (int)(t / Kpad), k = (int)(t - (long long)n * Kpad);
    out[t] = (k < K) ? (_Float16)in[(long long)k * N + n] : (_Float16)0.0f;
}

// ---------------------------------------------------------------------------
// WMMA GEMM: C[M,N](f32) = A[M,K](f16, row-major) * Bt[N,K](f16, col-major B)
// Block: 128 threads = 4 waves (wave32). Block tile 64x64, wave tile 32x32,
// each wave holds 2x2 accumulators of v_wmma_f32_16x16x32_f16.
// Both tiles live in LDS as [64 rows][32 k] stride 40 -> all fragment loads
// are contiguous ds_load_b128; interior blocks stage with unguarded
// global_load_b128 / ds_store_b128.
// ---------------------------------------------------------------------------
__global__ __launch_bounds__(128)
void gat_gemm_wmma(const _Float16* __restrict__ A, const _Float16* __restrict__ Bt,
                   float* __restrict__ C, int M, int K, int N)
{
    __shared__ _Float16 As[64][40];   // 64 M-rows x 32 K
    __shared__ _Float16 Bs[64][40];   // 64 N-cols x 32 K

    const int tid  = threadIdx.x;
    const int lane = tid & 31;
    const int wave = tid >> 5;        // 0..3
    const int wm   = wave >> 1;       // 0..1 (row sub-tile)
    const int wn   = wave & 1;        // 0..1 (col sub-tile)
    const int row0 = blockIdx.y * 64;
    const int col0 = blockIdx.x * 64;
    const int hsel = lane >> 4;       // 0 for lanes 0-15, 1 for 16-31

    const bool aFast = (row0 + 64 <= M) && ((K & 7) == 0);
    const bool bFast = (col0 + 64 <= N) && ((K & 7) == 0);

    // per-thread staging coordinates (2 x uint4 each): 64 rows x 4 chunks
    int r0 = tid >> 2,                 c0 = (tid & 3) << 3;
    int r1 = (128 + tid) >> 2,         c1 = ((128 + tid) & 3) << 3;
    const _Float16* aP0 = A  + (long long)(row0 + r0) * K + c0;
    const _Float16* aP1 = A  + (long long)(row0 + r1) * K + c1;
    const _Float16* bP0 = Bt + (long long)(col0 + r0) * K + c0;
    const _Float16* bP1 = Bt + (long long)(col0 + r1) * K + c1;

    v8f acc[2][2];
    #pragma unroll
    for (int a = 0; a < 2; ++a)
        #pragma unroll
        for (int b = 0; b < 2; ++b)
            acc[a][b] = (v8f)0.0f;

    const int ktiles = (K + 31) >> 5;
    for (int kt = 0; kt < ktiles; ++kt) {
        const int k0 = kt << 5;
        const bool kFull = (k0 + 32 <= K);

        // ---- stage A tile (64 rows x 32 halves) ----
        if (aFast && kFull) {
            *(uint4*)&As[r0][c0] = *(const uint4*)(aP0 + k0);
            *(uint4*)&As[r1][c1] = *(const uint4*)(aP1 + k0);
        } else {
            #pragma unroll
            for (int i = 0; i < 16; ++i) {
                int idx = i * 128 + tid;
                int r = idx >> 5, c = idx & 31;
                int gr = row0 + r, gc = k0 + c;
                As[r][c] = (gr < M && gc < K) ? A[(long long)gr * K + gc]
                                              : (_Float16)0.0f;
            }
        }
        // ---- stage B tile (64 cols x 32 halves) ----
        if (bFast && kFull) {
            *(uint4*)&Bs[r0][c0] = *(const uint4*)(bP0 + k0);
            *(uint4*)&Bs[r1][c1] = *(const uint4*)(bP1 + k0);
        } else {
            #pragma unroll
            for (int i = 0; i < 16; ++i) {
                int idx = i * 128 + tid;
                int r = idx >> 5, c = idx & 31;
                int gc = col0 + r, gk = k0 + c;
                Bs[r][c] = (gc < N && gk < K) ? Bt[(long long)gc * K + gk]
                                              : (_Float16)0.0f;
            }
        }
        // hint next k-tiles toward the caches (global_prefetch_b8)
        if (kt + 1 < ktiles) {
            __builtin_prefetch(aP0 + k0 + 32, 0, 1);
            __builtin_prefetch(bP0 + k0 + 32, 0, 1);
        }
        __syncthreads();

        // ---- fragments: all contiguous 16-half runs -> ds_load_b128 ----
        v16h afrag[2], bfrag[2];
        #pragma unroll
        for (int rm = 0; rm < 2; ++rm) {
            int arow = wm * 32 + rm * 16 + (lane & 15);
            #pragma unroll
            for (int j = 0; j < 16; ++j) {
                // A 16x32 f16: K = j + hsel*16 (contiguous, matches B)
                int kk = j + (hsel << 4);
                afrag[rm][j] = As[arow][kk];
            }
        }
        #pragma unroll
        for (int cn = 0; cn < 2; ++cn) {
            int bcol = wn * 32 + cn * 16 + (lane & 15);
            #pragma unroll
            for (int j = 0; j < 16; ++j) {
                // B 32x16 f16: K = j + hsel*16 (contiguous in transposed tile)
                int kk = j + (hsel << 4);
                bfrag[cn][j] = Bs[bcol][kk];
            }
        }
        #pragma unroll
        for (int rm = 0; rm < 2; ++rm)
            #pragma unroll
            for (int cn = 0; cn < 2; ++cn)
                acc[rm][cn] = __builtin_amdgcn_wmma_f32_16x16x32_f16(
                    false, afrag[rm], false, bfrag[cn],
                    (short)0, acc[rm][cn], false, false);
        __syncthreads();
    }

    // ---- epilogue: C 16x16 f32 layout: lanes 0-15 M=i, lanes 16-31 M=8+i ----
    #pragma unroll
    for (int rm = 0; rm < 2; ++rm)
        #pragma unroll
        for (int cn = 0; cn < 2; ++cn) {
            int colg = col0 + wn * 32 + cn * 16 + (lane & 15);
            #pragma unroll
            for (int i = 0; i < 8; ++i) {
                int rowg = row0 + wm * 32 + rm * 16 + i + (hsel << 3);
                if (rowg < M && colg < N)
                    C[(long long)rowg * N + colg] = acc[rm][cn][i];
            }
        }
}

// ---------------------------------------------------------------------------
// Helper device functions
// ---------------------------------------------------------------------------
__device__ __forceinline__ unsigned fkey(float f) {
    unsigned u = __float_as_uint(f);
    return (u & 0x80000000u) ? ~u : (u | 0x80000000u);
}
__device__ __forceinline__ float fdecode(unsigned k) {
    return (k & 0x80000000u) ? __uint_as_float(k & 0x7fffffffu)
                             : __uint_as_float(~k);
}
__device__ __forceinline__ int esrc(const int* ei, int e) {
    return (e < EE) ? ei[e] : (e - EE);
}
__device__ __forceinline__ int edst(const int* ei, int e) {
    return (e < EE) ? ei[EE + e] : (e - EE);
}

// ---------------------------------------------------------------------------
// Softmax / aggregation kernels (templated on heads, channels)
// ---------------------------------------------------------------------------
__global__ void zero_u32(unsigned* __restrict__ p, long long n) {
    long long t = (long long)blockIdx.x * blockDim.x + threadIdx.x;
    if (t < n) p[t] = 0u;
}

// per-node attention logits, one wave per (node,head) row with shfl reduction
template <int HT, int CT>
__global__ void node_scores(const float* __restrict__ h,
                            const float* __restrict__ asrc,
                            const float* __restrict__ adst,
                            float* __restrict__ s, float* __restrict__ d) {
    int row  = blockIdx.x * (blockDim.x >> 5) + (threadIdx.x >> 5);
    int lane = threadIdx.x & 31;
    if (row >= NN * HT) return;
    int node = row / HT, hd = row - node * HT;
    const float* hp = h + (long long)node * HT * CT + hd * CT;
    const float* as = asrc + hd * CT;
    const float* ad = adst + hd * CT;
    float ss = 0.f, dd = 0.f;
    for (int c = lane; c < CT; c += 32) {
        float v = hp[c];
        ss += v * as[c];
        dd += v * ad[c];
    }
    #pragma unroll
    for (int o = 16; o > 0; o >>= 1) {
        ss += __shfl_down(ss, o, 32);
        dd += __shfl_down(dd, o, 32);
    }
    if (lane == 0) { s[row] = ss; d[row] = dd; }
}

// e = leaky_relu(s[src]+d[dst]); segment max into mkeys (ordered-uint trick)
template <int HT>
__global__ void edge_scores(const int* __restrict__ ei,
                            const float* __restrict__ s, const float* __restrict__ d,
                            float* __restrict__ ebuf, unsigned* __restrict__ mkeys) {
    long long t = (long long)blockIdx.x * blockDim.x + threadIdx.x;
    if (t >= (long long)ETOT * HT) return;
    int e  = (int)(t / HT);
    int hd = (int)(t - (long long)e * HT);
    int si = esrc(ei, e), di = edst(ei, e);
    float v = s[si * HT + hd] + d[di * HT + hd];
    v = (v > 0.f) ? v : 0.2f * v;                 // leaky_relu(0.2)
    ebuf[t] = v;
    atomicMax(&mkeys[di * HT + hd], fkey(v));
}

// decode segment max; sanitize non-finite to 0 (matches reference)
__global__ void decode_max(const unsigned* __restrict__ mkeys,
                           float* __restrict__ mf, int n) {
    int t = blockIdx.x * blockDim.x + threadIdx.x;
    if (t >= n) return;
    float v = fdecode(mkeys[t]);
    mf[t] = isfinite(v) ? v : 0.0f;
}

// ex = exp(e - m[dst]); segment sum into z
template <int HT>
__global__ void edge_exp(const int* __restrict__ ei, float* __restrict__ ebuf,
                         const float* __restrict__ mf, float* __restrict__ z) {
    long long t = (long long)blockIdx.x * blockDim.x + threadIdx.x;
    if (t >= (long long)ETOT * HT) return;
    int e  = (int)(t / HT);
    int hd = (int)(t - (long long)e * HT);
    int di = edst(ei, e);
    float ex = expf(ebuf[t] - mf[di * HT + hd]);
    ebuf[t] = ex;
    atomicAdd(&z[di * HT + hd], ex);
}

// alpha = ex / (z[dst] + 1e-16)
template <int HT>
__global__ void edge_alpha(const int* __restrict__ ei,
                           float* __restrict__ ebuf, const float* __restrict__ z) {
    long long t = (long long)blockIdx.x * blockDim.x + threadIdx.x;
    if (t >= (long long)ETOT * HT) return;
    int e  = (int)(t / HT);
    int hd = (int)(t - (long long)e * HT);
    int di = edst(ei, e);
    ebuf[t] = ebuf[t] / (z[di * HT + hd] + 1e-16f);
}

// out[dst, c] += h[src, c] * alpha[e, head(c)]
template <int HT, int CT>
__global__ void edge_aggregate(const int* __restrict__ ei,
                               const float* __restrict__ alpha,
                               const float* __restrict__ h,
                               float* __restrict__ out) {
    constexpr int W = HT * CT;
    long long t = (long long)blockIdx.x * blockDim.x + threadIdx.x;
    if (t >= (long long)ETOT * W) return;
    int e = (int)(t / W);
    int c = (int)(t - (long long)e * W);
    int hd = c / CT;
    int si = esrc(ei, e), di = edst(ei, e);
    float a = alpha[(long long)e * HT + hd];
    atomicAdd(&out[(long long)di * W + c], h[(long long)si * W + c] * a);
}

// out_f16 = elu(in + bias)  -- activation feeds only the next GEMM
__global__ void bias_elu_f16(const float* __restrict__ in, const float* __restrict__ b,
                             _Float16* __restrict__ out, int n, int W) {
    long long t = (long long)blockIdx.x * blockDim.x + threadIdx.x;
    if (t >= (long long)n * W) return;
    int c = (int)(t % W);
    float v = in[t] + b[c];
    v = (v > 0.f) ? v : expm1f(v);
    out[t] = (_Float16)v;
}

// out_f32 = in + bias (final layer)
__global__ void bias_f32(const float* __restrict__ in, const float* __restrict__ b,
                         float* __restrict__ out, int n, int W) {
    long long t = (long long)blockIdx.x * blockDim.x + threadIdx.x;
    if (t >= (long long)n * W) return;
    int c = (int)(t % W);
    out[t] = in[t] + b[c];
}

// ---------------------------------------------------------------------------
// Host-side launcher
// ---------------------------------------------------------------------------
static inline int cdiv(long long a, long long b) { return (int)((a + b - 1) / b); }

template <int HT, int CT>
static void run_layer(const int* ei, const float* hbuf, float* aggbuf,
                      const float* as, const float* ad,
                      float* sbuf, float* dbuf, float* ebuf,
                      unsigned* mkeys, float* mf, float* zbuf,
                      hipStream_t stream) {
    const int BS = 256;
    long long nh = (long long)NN * HT;
    long long eh = (long long)ETOT * HT;
    long long nw = (long long)NN * HT * CT;
    long long ew = (long long)ETOT * HT * CT;
    node_scores<HT, CT><<<cdiv(nh * 32, BS), BS, 0, stream>>>(hbuf, as, ad, sbuf, dbuf);
    zero_u32<<<cdiv(nh, BS), BS, 0, stream>>>(mkeys, nh);
    zero_u32<<<cdiv(nh, BS), BS, 0, stream>>>((unsigned*)zbuf, nh);
    zero_u32<<<cdiv(nw, BS), BS, 0, stream>>>((unsigned*)aggbuf, nw);
    edge_scores<HT><<<cdiv(eh, BS), BS, 0, stream>>>(ei, sbuf, dbuf, ebuf, mkeys);
    decode_max<<<cdiv(nh, BS), BS, 0, stream>>>(mkeys, mf, (int)nh);
    edge_exp<HT><<<cdiv(eh, BS), BS, 0, stream>>>(ei, ebuf, mf, zbuf);
    edge_alpha<HT><<<cdiv(eh, BS), BS, 0, stream>>>(ei, ebuf, zbuf);
    edge_aggregate<HT, CT><<<cdiv(ew, BS), BS, 0, stream>>>(ei, ebuf, hbuf, aggbuf);
}

extern "C" void kernel_launch(void* const* d_in, const int* in_sizes, int n_in,
                              void* d_out, int out_size, void* d_ws, size_t ws_size,
                              hipStream_t stream) {
    const float* x    = (const float*)d_in[0];
    const int*   ei   = (const int*)  d_in[1];
    const float* W1   = (const float*)d_in[2];
    const float* a1s  = (const float*)d_in[3];
    const float* a1d  = (const float*)d_in[4];
    const float* b1   = (const float*)d_in[5];
    const float* W2   = (const float*)d_in[6];
    const float* a2s  = (const float*)d_in[7];
    const float* a2d  = (const float*)d_in[8];
    const float* b2   = (const float*)d_in[9];
    const float* W3   = (const float*)d_in[10];
    const float* a3s  = (const float*)d_in[11];
    const float* a3d  = (const float*)d_in[12];
    const float* b3   = (const float*)d_in[13];
    float* out = (float*)d_out;

    // ---- workspace layout ----
    float* ws = (float*)d_ws;
    float*    Hbuf  = ws;                           // N*512 f32 (gemm out)
    float*    AGG   = Hbuf + (long long)NN * WIDTH; // N*512 f32 (aggregation)
    float*    ebuf  = AGG  + (long long)NN * WIDTH; // ETOT*8 f32
    unsigned* mkeys = (unsigned*)(ebuf + (long long)ETOT * HEADS); // N*8
    float*    mf    = (float*)(mkeys + (long long)NN * HEADS);     // N*8
    float*    zbuf  = mf   + (long long)NN * HEADS;                // N*8
    float*    sbuf  = zbuf + (long long)NN * HEADS;                // N*8
    float*    dbuf  = sbuf + (long long)NN * HEADS;                // N*8
    _Float16* Ah    = (_Float16*)(dbuf + (long long)NN * HEADS);   // N*512 f16
    _Float16* xh    = Ah   + (long long)NN * WIDTH;                // N*64 f16 (padded)
    _Float16* W1t   = xh   + (long long)NN * KPAD1;                // 512*64 (B^T, padded)
    _Float16* W2t   = W1t  + (long long)WIDTH * KPAD1;             // 512*512 (B^T)
    _Float16* W3t   = W2t  + (long long)WIDTH * WIDTH;             // 121*512 (B^T)

    const int BS = 256;

    // ---- one-time conversions: x padded, weights transposed (+padded) ----
    cvt_pad_rows<<<cdiv((long long)NN * KPAD1, BS), BS, 0, stream>>>(x, xh, NN, F_IN, KPAD1);
    cvt_T_pad<<<cdiv((long long)WIDTH * KPAD1, BS), BS, 0, stream>>>(W1, W1t, F_IN, WIDTH, KPAD1);
    cvt_T_pad<<<cdiv((long long)WIDTH * WIDTH, BS), BS, 0, stream>>>(W2, W2t, WIDTH, WIDTH, WIDTH);
    cvt_T_pad<<<cdiv((long long)NCLS * WIDTH, BS), BS, 0, stream>>>(W3, W3t, WIDTH, NCLS, WIDTH);

    // ---------------- Layer 1: 50(->64 pad) -> (8 heads x 64), concat ------
    {
        dim3 g(cdiv(WIDTH, 64), cdiv(NN, 64));
        gat_gemm_wmma<<<g, 128, 0, stream>>>(xh, W1t, Hbuf, NN, KPAD1, WIDTH);
    }
    run_layer<HEADS, HID>(ei, Hbuf, AGG, a1s, a1d, sbuf, dbuf, ebuf, mkeys, mf, zbuf, stream);
    bias_elu_f16<<<cdiv((long long)NN * WIDTH, BS), BS, 0, stream>>>(AGG, b1, Ah, NN, WIDTH);

    // ---------------- Layer 2: 512 -> 512, 1 head ----------------
    {
        dim3 g(cdiv(WIDTH, 64), cdiv(NN, 64));
        gat_gemm_wmma<<<g, 128, 0, stream>>>(Ah, W2t, Hbuf, NN, WIDTH, WIDTH);
    }
    run_layer<1, WIDTH>(ei, Hbuf, AGG, a2s, a2d, sbuf, dbuf, ebuf, mkeys, mf, zbuf, stream);
    bias_elu_f16<<<cdiv((long long)NN * WIDTH, BS), BS, 0, stream>>>(AGG, b2, Ah, NN, WIDTH);

    // ---------------- Layer 3: 512 -> 121, 1 head, mean (= identity) -------
    {
        dim3 g(cdiv(NCLS, 64), cdiv(NN, 64));
        gat_gemm_wmma<<<g, 128, 0, stream>>>(Ah, W3t, Hbuf, NN, WIDTH, NCLS);
    }
    run_layer<1, NCLS>(ei, Hbuf, AGG, a3s, a3d, sbuf, dbuf, ebuf, mkeys, mf, zbuf, stream);
    bias_f32<<<cdiv((long long)NN * NCLS, BS), BS, 0, stream>>>(AGG, b3, out, NN, NCLS);
}
